// EncodeProcessDecode_30227979829770
// MI455X (gfx1250) — matrix operand order repaired and checked by reading, not compiled, not run on hardware
//
#include <hip/hip_runtime.h>
#include <hip/hip_bf16.h>
#include <stdint.h>

// ---------------------------------------------------------------------------
// CDNA5 (gfx1250) WMMA types
// ---------------------------------------------------------------------------
typedef __bf16 v16bf __attribute__((ext_vector_type(16)));
typedef float  v8f   __attribute__((ext_vector_type(8)));

__device__ __forceinline__ unsigned short f32_bf16(float f) {
  unsigned u = __float_as_uint(f);
  u += 0x7fffu + ((u >> 16) & 1u);           // round-to-nearest-even
  return (unsigned short)(u >> 16);
}
__device__ __forceinline__ float bf16_f32(unsigned short h) {
  return __uint_as_float(((unsigned)h) << 16);
}

// Segment of the (virtual) concatenated input matrix.
// mode: 0 = direct row, 1 = gather row via idx[], 2 = broadcast row 0
struct SegDesc { const unsigned short* ptr; const int* idx; int mode; int width; };

struct Mlp4Args {
  SegDesc seg[8];
  int nseg;
  int K0;                              // input width (multiple of 32)
  const unsigned short* W[4];          // bf16 weights, row-major [K][128]
  const float* Bias[4];                // fp32 biases [128]
  int M;                               // rows
  unsigned short* out;                 // bf16 output [M][128] (if !fuse_out)
  const float* wout;                   // fused 128->1 linear weight
  const float* bout;                   // fused bias (scalar)
  float* fout;                         // fused fp32 output [M]
  int fuse_out;
};

// ---------------------------------------------------------------------------
// Fused 4-layer MLP: out = L3(relu(L2(relu(L1(relu(L0(x)))))))
// Block = 128-row tile, 8 wave32 waves; each wave: 16 rows x 128 cols.
// Layer0 A-fragments gathered from global segments (segment-outer loop so
// gather rows/base pointers resolve once per segment); layers 1-3 stream
// activations through LDS (bf16). B fragments are software-pipelined one
// n-tile ahead so each load's latency hides under the previous WMMA.
// ---------------------------------------------------------------------------
__global__ __launch_bounds__(256) void mlp4_kernel(Mlp4Args a) {
  __shared__ unsigned short lds0[128 * 128];   // 32KB
  __shared__ unsigned short lds1[128 * 128];   // 32KB
  const int tid  = threadIdx.x;
  const int lane = tid & 31;
  const int wave = tid >> 5;
  const int lm   = lane & 15;     // A: M row / B,C,D: N col
  const int half = lane >> 4;     // K-half (A) / M-half (C,D)
  const int m0   = blockIdx.x * 128;

  v8f acc[8];

  auto zacc = [&]() {
    #pragma unroll
    for (int i = 0; i < 8; i++)
      #pragma unroll
      for (int j = 0; j < 8; j++) acc[i][j] = 0.0f;
  };
  // B fragment (32x16 bf16): lane -> N col (n0+lm), K = 8*half+s (s<8), 16+8*half+(s-8) (s>=8)
  auto loadB = [&](const unsigned short* W, int ks, int n0) -> v16bf {
    const unsigned short* p = W + (ks * 32 + 8 * half) * 128 + n0 + lm;
    union { unsigned short u[16]; v16bf v; } x;
    #pragma unroll
    for (int s = 0; s < 8; s++) {
      x.u[s]     = p[s * 128];
      x.u[s + 8] = p[(16 + s) * 128];
    }
    return x.v;
  };
  // One k-step: 8 WMMAs with B pipelined one n-tile ahead (load(nt+1) in
  // flight across wmma(nt); rotation forces the schedule).
  auto kstep_mma = [&](const unsigned short* W, int ks, v16bf af) {
    v16bf bc = loadB(W, ks, 0);
    #pragma unroll
    for (int nt = 0; nt < 8; nt++) {
      v16bf bn = bc;
      if (nt < 7) bn = loadB(W, ks, (nt + 1) * 16);
      acc[nt] = __builtin_amdgcn_wmma_f32_16x16x32_bf16(
          false, af, false, bc, (short)0, acc[nt], false, false);
      bc = bn;
    }
  };
  // Load a 16x32 A fragment given a per-row base pointer (already offset by
  // 8*half) and a k-offset within the segment.
  auto loadA = [&](const unsigned short* bp, int ko) -> v16bf {
    union { uint4 u[2]; v16bf v; } x;
    x.u[0] = *(const uint4*)(bp + ko);        // K = ko + 8*half .. +7
    x.u[1] = *(const uint4*)(bp + ko + 16);   // K = ko + 16 + 8*half .. +7
    return x.v;
  };

  // ---- layer 0: A from global segments (fused gather/concat) ----
  {
    zacc();
    int mrow = m0 + wave * 16 + lm;
    if (mrow >= a.M) mrow = a.M - 1;
    int kglob = 0;                      // running k-step index
    for (int s = 0; s < a.nseg; s++) {
      int row;
      if (a.seg[s].mode == 1)      row = a.seg[s].idx[mrow];
      else if (a.seg[s].mode == 2) row = 0;
      else                         row = mrow;
      const unsigned short* bp =
          a.seg[s].ptr + (long long)row * a.seg[s].width + 8 * half;
      const int nks = a.seg[s].width >> 5;
      for (int ks = 0; ks < nks; ks++)
        kstep_mma(a.W[0], kglob + ks, loadA(bp, ks * 32));
      kglob += nks;
    }
    #pragma unroll
    for (int nt = 0; nt < 8; nt++) {
      const int col = nt * 16 + lm;
      const float b = a.Bias[0][col];
      #pragma unroll
      for (int r = 0; r < 8; r++) {
        float v = acc[nt][r] + b;
        v = v > 0.f ? v : 0.f;
        lds0[(wave * 16 + r + 8 * half) * 128 + col] = f32_bf16(v);
      }
    }
  }
  __syncthreads();

  // ---- layers 1..3: A from LDS ----
  #pragma unroll
  for (int L = 1; L < 4; L++) {
    const unsigned short* src = (L == 2) ? lds1 : lds0;
    unsigned short*       dst = (L == 1) ? lds1 : lds0;
    zacc();
    const unsigned short* bp = src + (wave * 16 + lm) * 128 + 8 * half;
    #pragma unroll
    for (int ks = 0; ks < 4; ks++)
      kstep_mma(a.W[L], ks, loadA(bp, ks * 32));
    if (L < 3) {
      #pragma unroll
      for (int nt = 0; nt < 8; nt++) {
        const int col = nt * 16 + lm;
        const float b = a.Bias[L][col];
        #pragma unroll
        for (int r = 0; r < 8; r++) {
          float v = acc[nt][r] + b;
          v = v > 0.f ? v : 0.f;
          dst[(wave * 16 + r + 8 * half) * 128 + col] = f32_bf16(v);
        }
      }
      __syncthreads();
    } else if (!a.fuse_out) {
      #pragma unroll
      for (int nt = 0; nt < 8; nt++) {
        const int col = nt * 16 + lm;
        const float b = a.Bias[3][col];
        #pragma unroll
        for (int r = 0; r < 8; r++) {
          const int m = m0 + wave * 16 + r + 8 * half;
          if (m < a.M)
            a.out[(long long)m * 128 + col] = f32_bf16(acc[nt][r] + b);
        }
      }
    } else {
      // fused 128->1 output linear: per-row dot + 16-lane shuffle reduce
      float part[8];
      #pragma unroll
      for (int r = 0; r < 8; r++) part[r] = 0.f;
      #pragma unroll
      for (int nt = 0; nt < 8; nt++) {
        const int col = nt * 16 + lm;
        const float b  = a.Bias[3][col];
        const float wc = a.wout[col];
        #pragma unroll
        for (int r = 0; r < 8; r++) part[r] += (acc[nt][r] + b) * wc;
      }
      const float bo = a.bout[0];
      #pragma unroll
      for (int r = 0; r < 8; r++) {
        float s = part[r];
        s += __shfl_xor(s, 1, 32);
        s += __shfl_xor(s, 2, 32);
        s += __shfl_xor(s, 4, 32);
        s += __shfl_xor(s, 8, 32);
        if (lm == 0) {
          const int m = m0 + wave * 16 + r + 8 * half;
          if (m < a.M) a.fout[m] = s + bo;
        }
      }
    }
  }
}

// ---------------------------------------------------------------------------
// Helper kernels
// ---------------------------------------------------------------------------
__global__ void k_f32_to_bf16(const float* s, unsigned short* d, int n) {
  int i = blockIdx.x * blockDim.x + threadIdx.x;
  if (i < n) d[i] = f32_bf16(s[i]);
}
__global__ void k_zero_f32(float* p, int n) {
  int i = blockIdx.x * blockDim.x + threadIdx.x;
  if (i < n) p[i] = 0.f;
}
// agg[recv[e]] += e_new[e]  (fp32 atomics, bf16 source)
__global__ void k_segsum(const unsigned short* e, const int* recv, float* agg, int total) {
  int i = blockIdx.x * blockDim.x + threadIdx.x;
  if (i < total) {
    const int eid = i >> 7, c = i & 127;
    atomicAdd(&agg[recv[eid] * 128 + c], bf16_f32(e[i]));
  }
}
// out[0..127] += column sums of x[0..M)[128]  (128 threads/block, 256 rows/block)
__global__ void k_colsum(const unsigned short* x, float* out, int M) {
  const int col = threadIdx.x;
  int r0 = blockIdx.x * 256;
  int r1 = r0 + 256; if (r1 > M) r1 = M;
  float s = 0.f;
  for (int r = r0; r < r1; r++) s += bf16_f32(x[(long long)r * 128 + col]);
  atomicAdd(&out[col], s);
}

// Tiny fp32 MLP for the 1-row global feature (128 threads, 1 block)
struct GArgs {
  const float* in[4]; int win[4]; int nin;
  const float* W[4]; const float* B[4]; int K0;
  float* out;                              // fp32 [128] (may be null)
  const float* wout; const float* bout; float* fout; int fuse;
};
__global__ void gmlp_kernel(GArgs a) {
  __shared__ float x[512];
  __shared__ float y[128];
  const int t = threadIdx.x;               // 128 threads
  int off = 0;
  for (int i = 0; i < a.nin; i++) {
    for (int k = t; k < a.win[i]; k += 128) x[off + k] = a.in[i][k];
    off += a.win[i];
  }
  __syncthreads();
  int K = a.K0;
  for (int L = 0; L < 4; L++) {
    float s = a.B[L][t];
    for (int k = 0; k < K; k++) s += x[k] * a.W[L][(long long)k * 128 + t];
    if (L < 3) s = s > 0.f ? s : 0.f;
    __syncthreads();
    x[t] = s;
    __syncthreads();
    K = 128;
  }
  if (a.out) a.out[t] = x[t];
  if (a.fuse) {
    y[t] = x[t] * a.wout[t];
    __syncthreads();
    if (t == 0) {
      float s = 0.f;
      for (int i = 0; i < 128; i++) s += y[i];
      a.fout[0] = s + a.bout[0];
    }
  }
}

// ---------------------------------------------------------------------------
// Host driver
// ---------------------------------------------------------------------------
extern "C" void kernel_launch(void* const* d_in, const int* in_sizes, int n_in,
                              void* d_out, int out_size, void* d_ws, size_t ws_size,
                              hipStream_t stream) {
  const int N    = in_sizes[0] / 64;        // 10000
  const int E    = in_sizes[3];             // 160000
  const int Npad = ((N + 127) / 128) * 128;

  const float* nodes   = (const float*)d_in[0];
  const float* edges   = (const float*)d_in[1];
  const float* gglob   = (const float*)d_in[2];
  const int* senders   = (const int*)d_in[3];
  const int* receivers = (const int*)d_in[4];
  float* dout = (float*)d_out;
  auto F = [&](int i) { return (const float*)d_in[i]; };

  char* ws = (char*)d_ws;
  size_t off = 0;
  auto alloc = [&](size_t bytes) -> void* {
    void* p = ws + off;
    off = (off + bytes + 255) & ~(size_t)255;
    return p;
  };

  unsigned short* edges_b = (unsigned short*)alloc((size_t)E * 32 * 2);
  unsigned short* nodes_b = (unsigned short*)alloc((size_t)N * 64 * 2);
  unsigned short* e0 = (unsigned short*)alloc((size_t)E * 128 * 2);
  unsigned short* e1 = (unsigned short*)alloc((size_t)E * 128 * 2);
  unsigned short* e2 = (unsigned short*)alloc((size_t)E * 128 * 2);
  unsigned short* n0 = (unsigned short*)alloc((size_t)Npad * 128 * 2);
  unsigned short* n1 = (unsigned short*)alloc((size_t)Npad * 128 * 2);
  unsigned short* n2 = (unsigned short*)alloc((size_t)Npad * 128 * 2);
  float*          agg_f = (float*)alloc((size_t)Npad * 128 * 4);
  unsigned short* agg_b = (unsigned short*)alloc((size_t)Npad * 128 * 2);
  float* sums = (float*)alloc(256 * 4);              // [sum_e | sum_n]
  float* g0f  = (float*)alloc(128 * 4);
  float* g1f  = (float*)alloc(128 * 4);
  float* g2f  = (float*)alloc(128 * 4);
  unsigned short* g0b = (unsigned short*)alloc(128 * 2);
  unsigned short* g1b = (unsigned short*)alloc(128 * 2);

  // --- bf16 weight conversion (weights stay L2-resident; 1.5 MB total) ---
  struct WS { int idx; int K; };
  static const WS wspec[32] = {
    {5,32},{7,128},{9,128},{11,128},       // enc edge
    {13,64},{15,128},{17,128},{19,128},    // enc node
    {29,512},{31,128},{33,128},{35,128},   // blk1 edge
    {37,384},{39,128},{41,128},{43,128},   // blk1 node
    {53,1024},{55,128},{57,128},{59,128},  // blk2 edge
    {61,640},{63,128},{65,128},{67,128},   // blk2 node
    {77,128},{79,128},{81,128},{83,128},   // dec edge
    {85,128},{87,128},{89,128},{91,128},   // dec node
  };
  unsigned short* wbf[32];
  for (int i = 0; i < 32; i++) {
    const int n = wspec[i].K * 128;
    wbf[i] = (unsigned short*)alloc((size_t)n * 2);
    k_f32_to_bf16<<<(n + 255) / 256, 256, 0, stream>>>(F(wspec[i].idx), wbf[i], n);
  }
  k_f32_to_bf16<<<(E * 32 + 255) / 256, 256, 0, stream>>>(edges, edges_b, E * 32);
  k_f32_to_bf16<<<(N * 64 + 255) / 256, 256, 0, stream>>>(nodes, nodes_b, N * 64);

  auto mk_seg = [](const unsigned short* p, const int* idx, int mode, int w) {
    SegDesc s; s.ptr = p; s.idx = idx; s.mode = mode; s.width = w; return s;
  };
  auto run = [&](const SegDesc* segs, int nseg, int K0, unsigned short* const* Wd,
                 const int* bidx, int M, unsigned short* outp,
                 const float* wo, const float* bo, float* fo) {
    Mlp4Args a{};
    for (int i = 0; i < nseg; i++) a.seg[i] = segs[i];
    a.nseg = nseg; a.K0 = K0;
    for (int l = 0; l < 4; l++) { a.W[l] = Wd[l]; a.Bias[l] = F(bidx[l]); }
    a.M = M; a.out = outp; a.wout = wo; a.bout = bo; a.fout = fo;
    a.fuse_out = (fo != nullptr);
    mlp4_kernel<<<(M + 127) / 128, 256, 0, stream>>>(a);
  };
  auto run_g = [&](const float* const* ins, const int* wins, int nin, int K0,
                   const int* widx, float* outp, const float* wo, const float* bo,
                   float* fo) {
    GArgs g{};
    for (int i = 0; i < nin; i++) { g.in[i] = ins[i]; g.win[i] = wins[i]; }
    g.nin = nin; g.K0 = K0;
    for (int l = 0; l < 4; l++) { g.W[l] = F(widx[l]); g.B[l] = F(widx[l] + 1); }
    g.out = outp; g.wout = wo; g.bout = bo; g.fout = fo; g.fuse = (fo != nullptr);
    gmlp_kernel<<<1, 128, 0, stream>>>(g);
  };

  // ---------------- encoder ----------------
  { SegDesc s[1] = { mk_seg(edges_b, nullptr, 0, 32) };
    unsigned short* W[4] = { wbf[0], wbf[1], wbf[2], wbf[3] };
    int b[4] = {6, 8, 10, 12};
    run(s, 1, 32, W, b, E, e0, nullptr, nullptr, nullptr); }
  { SegDesc s[1] = { mk_seg(nodes_b, nullptr, 0, 64) };
    unsigned short* W[4] = { wbf[4], wbf[5], wbf[6], wbf[7] };
    int b[4] = {14, 16, 18, 20};
    run(s, 1, 64, W, b, N, n0, nullptr, nullptr, nullptr); }
  { const float* ins[1] = { gglob }; int wins[1] = { 16 };
    int widx[4] = {21, 23, 25, 27};
    run_g(ins, wins, 1, 16, widx, g0f, nullptr, nullptr, nullptr); }
  k_f32_to_bf16<<<1, 128, 0, stream>>>(g0f, g0b, 128);

  // ---------------- core block 1 ----------------
  { SegDesc s[4] = { mk_seg(e0, nullptr, 0, 128), mk_seg(n0, receivers, 1, 128),
                     mk_seg(n0, senders, 1, 128), mk_seg(g0b, nullptr, 2, 128) };
    unsigned short* W[4] = { wbf[8], wbf[9], wbf[10], wbf[11] };
    int b[4] = {30, 32, 34, 36};
    run(s, 4, 512, W, b, E, e1, nullptr, nullptr, nullptr); }
  k_zero_f32<<<(Npad * 128 + 255) / 256, 256, 0, stream>>>(agg_f, Npad * 128);
  k_segsum<<<(E * 128 + 255) / 256, 256, 0, stream>>>(e1, receivers, agg_f, E * 128);
  k_f32_to_bf16<<<(Npad * 128 + 255) / 256, 256, 0, stream>>>(agg_f, agg_b, Npad * 128);
  { SegDesc s[3] = { mk_seg(agg_b, nullptr, 0, 128), mk_seg(n0, nullptr, 0, 128),
                     mk_seg(g0b, nullptr, 2, 128) };
    unsigned short* W[4] = { wbf[12], wbf[13], wbf[14], wbf[15] };
    int b[4] = {38, 40, 42, 44};
    run(s, 3, 384, W, b, N, n1, nullptr, nullptr, nullptr); }
  k_zero_f32<<<1, 256, 0, stream>>>(sums, 256);
  k_colsum<<<(E + 255) / 256, 128, 0, stream>>>(e1, sums, E);
  k_colsum<<<(N + 255) / 256, 128, 0, stream>>>(n1, sums + 128, N);
  { const float* ins[3] = { sums, sums + 128, g0f }; int wins[3] = {128, 128, 128};
    int widx[4] = {45, 47, 49, 51};
    run_g(ins, wins, 3, 384, widx, g1f, nullptr, nullptr, nullptr); }
  k_f32_to_bf16<<<1, 128, 0, stream>>>(g1f, g1b, 128);

  // ---------------- core block 2 (graph_concat doubled widths) ----------------
  { SegDesc s[8] = { mk_seg(e0, nullptr, 0, 128), mk_seg(e1, nullptr, 0, 128),
                     mk_seg(n0, receivers, 1, 128), mk_seg(n1, receivers, 1, 128),
                     mk_seg(n0, senders, 1, 128),   mk_seg(n1, senders, 1, 128),
                     mk_seg(g0b, nullptr, 2, 128),  mk_seg(g1b, nullptr, 2, 128) };
    unsigned short* W[4] = { wbf[16], wbf[17], wbf[18], wbf[19] };
    int b[4] = {54, 56, 58, 60};
    run(s, 8, 1024, W, b, E, e2, nullptr, nullptr, nullptr); }
  k_zero_f32<<<(Npad * 128 + 255) / 256, 256, 0, stream>>>(agg_f, Npad * 128);
  k_segsum<<<(E * 128 + 255) / 256, 256, 0, stream>>>(e2, receivers, agg_f, E * 128);
  k_f32_to_bf16<<<(Npad * 128 + 255) / 256, 256, 0, stream>>>(agg_f, agg_b, Npad * 128);
  { SegDesc s[5] = { mk_seg(agg_b, nullptr, 0, 128), mk_seg(n0, nullptr, 0, 128),
                     mk_seg(n1, nullptr, 0, 128),    mk_seg(g0b, nullptr, 2, 128),
                     mk_seg(g1b, nullptr, 2, 128) };
    unsigned short* W[4] = { wbf[20], wbf[21], wbf[22], wbf[23] };
    int b[4] = {62, 64, 66, 68};
    run(s, 5, 640, W, b, N, n2, nullptr, nullptr, nullptr); }
  k_zero_f32<<<1, 256, 0, stream>>>(sums, 256);
  k_colsum<<<(E + 255) / 256, 128, 0, stream>>>(e2, sums, E);
  k_colsum<<<(N + 255) / 256, 128, 0, stream>>>(n2, sums + 128, N);
  { const float* ins[4] = { sums, sums + 128, g0f, g1f };
    int wins[4] = {128, 128, 128, 128};
    int widx[4] = {69, 71, 73, 75};
    run_g(ins, wins, 4, 512, widx, g2f, nullptr, nullptr, nullptr); }

  // ---------------- decoder + fused 128->1 output linears ----------------
  { SegDesc s[1] = { mk_seg(e2, nullptr, 0, 128) };
    unsigned short* W[4] = { wbf[24], wbf[25], wbf[26], wbf[27] };
    int b[4] = {78, 80, 82, 84};
    run(s, 1, 128, W, b, E, nullptr, F(101), F(102), dout); }              // e_out
  { SegDesc s[1] = { mk_seg(n2, nullptr, 0, 128) };
    unsigned short* W[4] = { wbf[28], wbf[29], wbf[30], wbf[31] };
    int b[4] = {86, 88, 90, 92};
    run(s, 1, 128, W, b, N, nullptr, F(103), F(104), dout + E); }          // n_out
  { const float* ins[1] = { g2f }; int wins[1] = { 128 };
    int widx[4] = {93, 95, 97, 99};
    run_g(ins, wins, 1, 128, widx, nullptr, F(105), F(106), dout + E + N); } // g_out

  (void)n_in; (void)out_size; (void)ws_size;
}
